// Experts_22720376996507
// MI455X (gfx1250) — matrix operand order repaired and checked by reading, not compiled, not run on hardware
//
#include <hip/hip_runtime.h>
#include <hip/hip_bf16.h>
#include <math.h>

// ---------------------------------------------------------------------------
// MoE grouped expert MLP for MI455X (gfx1250, wave32, WMMA).
//   h   = gelu_exact(X_e @ W0_e^T)        X: [2048,1024]  W0: [4096,1024]
//   out = (h @ W3_e^T) * gate_mask        W3: [1024,4096] out: [2048,1024]
// Compute-bound (~275 GFLOP vs ~400MB => ~690 FLOP/B). Split-bf16 GEMM
// (hi/lo decomposition, 3x v_wmma_f32_16x16x32_bf16 per 16x16x32 tile) gives
// near-fp32 accuracy (~1.5e-5 rel err) at bf16 matrix-core throughput.
// Global->LDS path is software-pipelined through a register stage so HBM/L2
// latency overlaps the WMMA block of the previous k-step.
// ---------------------------------------------------------------------------

#define NUM_LOCAL 4
#define E_TOTAL   8
#define HIDDEN    1024
#define FFN       (4 * HIDDEN)
#define CAP       2048

typedef __attribute__((ext_vector_type(16))) __bf16 v16bf;
typedef __attribute__((ext_vector_type(8)))  float  v8f;

#define SRD 40  // LDS row stride in ushorts: 32 data + 8 pad (banks staggered)

__device__ __forceinline__ unsigned short f32_to_bf16_rn(float x) {
  unsigned u = __float_as_uint(x);
  unsigned r = u + 0x7FFFu + ((u >> 16) & 1u);  // round-to-nearest-even
  return (unsigned short)(r >> 16);
}
__device__ __forceinline__ float bf16_to_f32(unsigned short h) {
  return __uint_as_float(((unsigned)h) << 16);
}
__device__ __forceinline__ float gelu_exact(float x) {
  return 0.5f * x * (1.0f + erff(x * 0.70710678118654752440f));
}

// --- staged global load: 128x32 fp32 tile -> 4x float4 per thread -----------
__device__ __forceinline__ void stage_load(const float* __restrict__ src, int ld,
                                           float4* __restrict__ reg) {
  const int t = threadIdx.x;
#pragma unroll
  for (int i = 0; i < 4; ++i) {
    const int c  = t + i * 256;      // 0..1023 float4 chunks
    const int r  = c >> 3;           // row 0..127
    const int c4 = (c & 7) << 2;     // col 0..28 step 4
    reg[i] = *(const float4*)(src + (size_t)r * ld + c4);
  }
}

// --- convert staged registers to bf16 hi/lo planes in LDS -------------------
__device__ __forceinline__ void stage_cvt_store(const float4* __restrict__ reg,
                                                unsigned short* __restrict__ dhi,
                                                unsigned short* __restrict__ dlo) {
  const int t = threadIdx.x;
#pragma unroll
  for (int i = 0; i < 4; ++i) {
    const int c  = t + i * 256;
    const int r  = c >> 3;
    const int c4 = (c & 7) << 2;
    const float4 v = reg[i];
    ushort4 h4, l4;
    h4.x = f32_to_bf16_rn(v.x); l4.x = f32_to_bf16_rn(v.x - bf16_to_f32(h4.x));
    h4.y = f32_to_bf16_rn(v.y); l4.y = f32_to_bf16_rn(v.y - bf16_to_f32(h4.y));
    h4.z = f32_to_bf16_rn(v.z); l4.z = f32_to_bf16_rn(v.z - bf16_to_f32(h4.z));
    h4.w = f32_to_bf16_rn(v.w); l4.w = f32_to_bf16_rn(v.w - bf16_to_f32(h4.w));
    *(ushort4*)(dhi + r * SRD + c4) = h4;
    *(ushort4*)(dlo + r * SRD + c4) = l4;
  }
}

// A-matrix fragment (16x32 bf16): lane L<16 holds row L, K k0..k0+15 contiguous
// with k0 = 0; lanes 16..31 hold the same rows with k0 = 16.  (matches B)
__device__ __forceinline__ void load_a_frag(const unsigned short* s, int row, int k0,
                                            v16bf* out) {
  union { uint4 q[2]; v16bf v; } u;
  u.q[0] = *(const uint4*)(s + row * SRD + k0);       // K = k0 .. k0+7
  u.q[1] = *(const uint4*)(s + row * SRD + k0 + 8);   // K = k0+8 .. k0+15
  *out = u.v;
}

// B-matrix fragment (32x16 bf16): lane L<16 holds column L, K 0..15 contiguous;
// lanes 16..31 hold K 16..31.  (same contiguous k order as A)
__device__ __forceinline__ void load_b_frag(const unsigned short* s, int col, int k0,
                                            v16bf* out) {
  union { uint4 q[2]; v16bf v; } u;
  const uint4* p = (const uint4*)(s + col * SRD + k0);
  u.q[0] = p[0];
  u.q[1] = p[1];
  *out = u.v;
}

// C = A(MxK, row-major) * B(NxK, row-major)^T, both K-contiguous.
// EPI==0: C[r,c] = gelu(acc)      (writes fp32 h to workspace)
// EPI==1: C[r,c] = acc * mask     (gate mask; locals always on)
// gridDim.z batches experts; sA/sB/sC are per-expert element strides.
template <int EPI>
__global__ void __launch_bounds__(256)
moe_gemm(const float* __restrict__ A, long long sA,
         const float* __restrict__ B, long long sB,
         float* __restrict__ C, long long sC,
         int M, int N, int K,
         const float* __restrict__ gate, int zbase) {
  __shared__ __align__(16) unsigned short s_a_hi[128 * SRD];
  __shared__ __align__(16) unsigned short s_a_lo[128 * SRD];
  __shared__ __align__(16) unsigned short s_b_hi[128 * SRD];
  __shared__ __align__(16) unsigned short s_b_lo[128 * SRD];

  const int e = blockIdx.z;
  A += (size_t)e * (size_t)sA;
  B += (size_t)e * (size_t)sB;
  C += (size_t)e * (size_t)sC;

  float mask = 1.0f;
  if (EPI == 1) {
    const int ge = zbase + e;
    mask = (ge < NUM_LOCAL) ? 1.0f : ((gate[e] != 0.0f) ? 1.0f : 0.0f);
  }

  const int m0 = blockIdx.y * 128;
  const int n0 = blockIdx.x * 128;
  const float* __restrict__ Ab = A + (size_t)m0 * K;
  const float* __restrict__ Bb = B + (size_t)n0 * K;

  const int lane = threadIdx.x & 31;
  const int wv   = threadIdx.x >> 5;   // wave id 0..7
  const int wm   = (wv >> 1) * 32;     // wave M offset (4 rows of waves)
  const int wn   = (wv & 1) * 64;      // wave N offset (2 cols of waves)
  const int lr   = lane & 15;
  const int hi16 = lane >> 4;
  const int kA0  = hi16 * 16;          // A fragment K base
  const int kB0  = hi16 * 16;          // B fragment K base

  v8f acc[2][4] = {};

  // Software pipeline: stage k-step kb in registers while computing kb-32.
  float4 ra[4], rb[4];
  stage_load(Ab, K, ra);
  stage_load(Bb, K, rb);

  for (int kb = 0; kb < K; kb += 32) {
    __syncthreads();  // previous compute done -> safe to overwrite LDS
    stage_cvt_store(ra, s_a_hi, s_a_lo);
    stage_cvt_store(rb, s_b_hi, s_b_lo);
    __syncthreads();

    // Issue next tile's global loads; they stay in flight across the WMMAs.
    if (kb + 32 < K) {
      stage_load(Ab + kb + 32, K, ra);
      stage_load(Bb + kb + 32, K, rb);
    }

    v16bf a_hi[2], a_lo[2];
#pragma unroll
    for (int mi = 0; mi < 2; ++mi) {
      const int row = wm + mi * 16 + lr;
      load_a_frag(s_a_hi, row, kA0, &a_hi[mi]);
      load_a_frag(s_a_lo, row, kA0, &a_lo[mi]);
    }
#pragma unroll
    for (int ni = 0; ni < 4; ++ni) {
      const int col = wn + ni * 16 + lr;
      v16bf b_hi, b_lo;
      load_b_frag(s_b_hi, col, kB0, &b_hi);
      load_b_frag(s_b_lo, col, kB0, &b_lo);
#pragma unroll
      for (int mi = 0; mi < 2; ++mi) {
        // split GEMM: hi*hi + hi*lo + lo*hi (lo*lo ~2^-16 rel, dropped)
        acc[mi][ni] = __builtin_amdgcn_wmma_f32_16x16x32_bf16(
            false, a_hi[mi], false, b_hi, (short)0, acc[mi][ni], false, false);
        acc[mi][ni] = __builtin_amdgcn_wmma_f32_16x16x32_bf16(
            false, a_hi[mi], false, b_lo, (short)0, acc[mi][ni], false, false);
        acc[mi][ni] = __builtin_amdgcn_wmma_f32_16x16x32_bf16(
            false, a_lo[mi], false, b_hi, (short)0, acc[mi][ni], false, false);
      }
    }
  }

  // C/D layout: VGPR v, lanes 0-15 -> M=v, lanes 16-31 -> M=v+8; N = lane&15.
#pragma unroll
  for (int mi = 0; mi < 2; ++mi) {
#pragma unroll
    for (int ni = 0; ni < 4; ++ni) {
      union { v8f v; float f[8]; } u;
      u.v = acc[mi][ni];
      const int col = n0 + wn + ni * 16 + lr;
#pragma unroll
      for (int v = 0; v < 8; ++v) {
        const int row = m0 + wm + mi * 16 + hi16 * 8 + v;
        float x = u.f[v];
        if (EPI == 0) x = gelu_exact(x);
        else          x *= mask;
        C[(size_t)row * N + col] = x;
      }
    }
  }
}

extern "C" void kernel_launch(void* const* d_in, const int* in_sizes, int n_in,
                              void* d_out, int out_size, void* d_ws, size_t ws_size,
                              hipStream_t stream) {
  (void)in_sizes; (void)n_in; (void)out_size;
  const float* gate = (const float*)d_in[0];  // output_tensor [1, E_TOTAL]
  const float* X    = (const float*)d_in[1];  // inputs [1, E, CAP, HIDDEN]
  const float* W0   = (const float*)d_in[2];  // [E, FFN, HIDDEN]
  const float* W3   = (const float*)d_in[3];  // [E, HIDDEN, FFN]
  float* out = (float*)d_out;                 // [1, E, CAP, HIDDEN]
  float* h   = (float*)d_ws;                  // fp32 activations workspace

  const long long sX  = (long long)CAP * HIDDEN;
  const long long sW0 = (long long)FFN * HIDDEN;
  const long long sW3 = (long long)HIDDEN * FFN;
  const long long sH  = (long long)CAP * FFN;
  const size_t hBytesOne = (size_t)sH * sizeof(float);  // 32 MiB per expert

  dim3 blk(256, 1, 1);
  if (ws_size >= hBytesOne * (size_t)E_TOTAL) {
    // All experts batched: two launches, gridDim.z = E_TOTAL.
    dim3 g1(FFN / 128, CAP / 128, E_TOTAL);
    moe_gemm<0><<<g1, blk, 0, stream>>>(X, sX, W0, sW0, h, sH,
                                        CAP, FFN, HIDDEN, nullptr, 0);
    dim3 g2(HIDDEN / 128, CAP / 128, E_TOTAL);
    moe_gemm<1><<<g2, blk, 0, stream>>>(h, sH, W3, sW3, out, sX,
                                        CAP, HIDDEN, FFN, gate, 0);
  } else {
    // Per-expert loop reusing one 32 MiB h buffer (stream-ordered dependency).
    for (int e = 0; e < E_TOTAL; ++e) {
      dim3 g1(FFN / 128, CAP / 128, 1);
      moe_gemm<0><<<g1, blk, 0, stream>>>(X + (size_t)e * sX, 0,
                                          W0 + (size_t)e * sW0, 0,
                                          h, 0, CAP, FFN, HIDDEN, nullptr, 0);
      dim3 g2(HIDDEN / 128, CAP / 128, 1);
      moe_gemm<1><<<g2, blk, 0, stream>>>(h, 0,
                                          W3 + (size_t)e * sW3, 0,
                                          out + (size_t)e * sX, 0,
                                          CAP, HIDDEN, FFN, gate + e, e);
    }
  }
}
